// CSwinAttention_66005057405326
// MI455X (gfx1250) — matrix operand order repaired and checked by reading, not compiled, not run on hardware
//
#include <hip/hip_runtime.h>
#include <hip/hip_bf16.h>

// ---------------- types ----------------
typedef __attribute__((ext_vector_type(16))) __bf16          v16bf;
typedef __attribute__((ext_vector_type(8)))  float           v8f;
typedef __attribute__((ext_vector_type(16))) unsigned short  ushort16;
typedef __attribute__((ext_vector_type(8)))  unsigned short  ushort8;
typedef unsigned short u16;

union V16 {
    ushort16 u;
    ushort8  h[2];
    v16bf    b;
};

// fp32 -> bf16 (round to nearest even)
__device__ __forceinline__ u16 f2bf(float f) {
    unsigned int x = __builtin_bit_cast(unsigned int, f);
    unsigned int r = x + 0x7FFFu + ((x >> 16) & 1u);
    return (u16)(r >> 16);
}
__device__ __forceinline__ float bf2f(u16 h) {
    unsigned int x = ((unsigned int)h) << 16;
    return __builtin_bit_cast(float, x);
}

// ---------------- gfx1250 async global->LDS copy (ASYNCcnt path) ----------
// Per-lane: copies 16 bytes from global address gaddr to LDS byte address lds_byte_addr.
__device__ __forceinline__ void async_copy_b128(unsigned lds_byte_addr,
                                                const void* gaddr) {
    asm volatile("global_load_async_to_lds_b128 %0, %1, off"
                 :: "v"(lds_byte_addr), "v"(gaddr)
                 : "memory");
}
__device__ __forceinline__ void wait_asynccnt0() {
    asm volatile("s_wait_asynccnt 0x0" ::: "memory");
}
// Generic shared pointer -> LDS byte offset (aperture maps addr[31:0] to LDS).
__device__ __forceinline__ unsigned lds_addr_of(const void* p) {
    return (unsigned)(unsigned long long)p;
}

// ---------------- problem constants ----------------
#define BB   8
#define HH   64
#define WW   64
#define CC   512
#define LL   (HH * WW)        // 4096
#define MTOT (BB * LL)        // 32768
#define HALF (CC / 2)         // 256
#define NH   8                // heads per half
#define HD   32               // head dim
#define SS   8                // stripe size
#define NS   8                // HH/SS
#define NQKV (3 * CC)         // 1536

// ---------------- fp32 -> bf16 convert (8 elems / thread) ----------------
__global__ __launch_bounds__(256)
void cvt_f32_bf16x8_kernel(const float* __restrict__ src,
                           u16* __restrict__ dst, int n8) {
    int i = blockIdx.x * blockDim.x + threadIdx.x;
    if (i >= n8) return;
    const float4 f0 = *(const float4*)(src + (size_t)i * 8);
    const float4 f1 = *(const float4*)(src + (size_t)i * 8 + 4);
    ushort8 o;
    o[0] = f2bf(f0.x); o[1] = f2bf(f0.y); o[2] = f2bf(f0.z); o[3] = f2bf(f0.w);
    o[4] = f2bf(f1.x); o[5] = f2bf(f1.y); o[6] = f2bf(f1.z); o[7] = f2bf(f1.w);
    *(ushort8*)(dst + (size_t)i * 8) = o;
}

// ---------------- async double-buffered LDS WMMA GEMM ----------------
// C[m,n] = sum_k A[m,k] * Bw[n,k] (+ bias[n]); A,Bw bf16 row-major.
// Block: 256 threads = 8 waves (2 M x 4 N), tile 64M x 128N, K-step 32.
// Each wave -> 32x32 (2x2 WMMA 16x16x32 bf16 frags, f32 accumulate).
// Staging via GLOBAL_LOAD_ASYNC_TO_LDS_B128 with 2-deep LDS pipeline:
// one s_wait_asynccnt + one barrier per K-step, DMA overlaps WMMA.
#define LDS_STRIDE 40   // halfs per row: 80B (16B aligned, 20-bank skew)

template <bool OUT_BF16>
__global__ __launch_bounds__(256)
void wmma_gemm_async_kernel(const u16* __restrict__ Abf,
                            const u16* __restrict__ Bw,
                            const float* __restrict__ bias,
                            void* __restrict__ Out,
                            int M, int N, int K) {
    __shared__ u16 As[2][64 * LDS_STRIDE];    // 2 x  5,120 B
    __shared__ u16 Bs[2][128 * LDS_STRIDE];   // 2 x 10,240 B

    const int tid   = threadIdx.x;
    const int wave  = tid >> 5;
    const int lane  = tid & 31;
    const int wm    = wave >> 2;          // 0..1
    const int wn    = wave & 3;           // 0..3
    const int mblk  = blockIdx.x * 64;
    const int nblk  = blockIdx.y * 128;
    const int lrow  = lane & 15;
    const int khalf = lane >> 4;          // 0/1

    // cooperative DMA assignments: A = 16B/thread, B = 32B/thread (2 ops)
    const int arow = tid >> 2;            // 0..63
    const int acol = (tid & 3) * 8;       // 0/8/16/24
    const int brow = tid >> 1;            // 0..127
    const int bcol = (tid & 1) * 16;      // 0/16

    const u16* agp = Abf + (size_t)(mblk + arow) * K + acol;
    const u16* bgp = Bw  + (size_t)(nblk + brow) * K + bcol;

    unsigned as_off[2], bs_off[2];
    #pragma unroll
    for (int p = 0; p < 2; p++) {
        as_off[p] = lds_addr_of(&As[p][arow * LDS_STRIDE + acol]);
        bs_off[p] = lds_addr_of(&Bs[p][brow * LDS_STRIDE + bcol]);
    }

    v8f cacc[2][2];
    #pragma unroll
    for (int a = 0; a < 2; a++)
        #pragma unroll
        for (int b = 0; b < 2; b++)
            #pragma unroll
            for (int i = 0; i < 8; i++)
                cacc[a][b][i] = 0.0f;

    const int NIT = K >> 5;

    // prologue: DMA tile 0 into buffer 0
    async_copy_b128(as_off[0], agp);
    async_copy_b128(bs_off[0], bgp);
    async_copy_b128(bs_off[0] + 16, bgp + 8);

    for (int it = 0; it < NIT; ++it) {
        const int cur = it & 1;

        wait_asynccnt0();      // our DMA for tile `it` has landed in LDS
        __syncthreads();       // everyone's has; prior readers of buf are done

        // ---- fragments from LDS (ISA wave32 16-bit A/B layouts) ----
        V16 afrag[2], bfrag[2];
        #pragma unroll
        for (int mi = 0; mi < 2; mi++) {
            const int r = (wm * 32 + mi * 16 + lrow) * LDS_STRIDE;
            afrag[mi].h[0] = *(const ushort8*)&As[cur][r + khalf * 8];
            afrag[mi].h[1] = *(const ushort8*)&As[cur][r + 16 + khalf * 8];
        }
        #pragma unroll
        for (int ni = 0; ni < 2; ni++) {
            const int r = (wn * 32 + ni * 16 + lrow) * LDS_STRIDE;
            bfrag[ni].h[0] = *(const ushort8*)&Bs[cur][r + khalf * 8];
            bfrag[ni].h[1] = *(const ushort8*)&Bs[cur][r + 16 + khalf * 8];
        }

        // ---- 4 WMMA ops ----
        #pragma unroll
        for (int mi = 0; mi < 2; mi++)
            #pragma unroll
            for (int ni = 0; ni < 2; ni++)
                cacc[mi][ni] = __builtin_amdgcn_wmma_f32_16x16x32_bf16(
                    false, afrag[mi].b, false, bfrag[ni].b,
                    (short)0, cacc[mi][ni], false, false);

        // ---- issue DMA for next tile into the other buffer ----
        if (it + 1 < NIT) {
            const int k0  = (it + 1) << 5;
            const int nxt = cur ^ 1;
            async_copy_b128(as_off[nxt], agp + k0);
            async_copy_b128(bs_off[nxt], bgp + k0);
            async_copy_b128(bs_off[nxt] + 16, bgp + k0 + 8);
        }
    }

    // ---- store: C/D layout — VGPR i holds (M = i + 8*khalf, N = lane&15) ----
    #pragma unroll
    for (int mi = 0; mi < 2; mi++) {
        #pragma unroll
        for (int ni = 0; ni < 2; ni++) {
            const int n  = nblk + wn * 32 + ni * 16 + lrow;
            const int mb = mblk + wm * 32 + mi * 16 + khalf * 8;
            const float bv = (bias != nullptr) ? bias[n] : 0.0f;
            #pragma unroll
            for (int i = 0; i < 8; i++) {
                float v = cacc[mi][ni][i] + bv;
                if (OUT_BF16)
                    ((u16*)Out)[(size_t)(mb + i) * N + n] = f2bf(v);
                else
                    ((float*)Out)[(size_t)(mb + i) * N + n] = v;
            }
        }
    }
}

// ---------------- fused stripe attention + LePE ----------------
__device__ __forceinline__ void load_row32(const u16* __restrict__ p,
                                           float* __restrict__ dst) {
    ushort16 u0 = *(const ushort16*)p;
    ushort16 u1 = *(const ushort16*)(p + 16);
    #pragma unroll
    for (int i = 0; i < 16; i++) {
        dst[i]      = bf2f(u0[i]);
        dst[16 + i] = bf2f(u1[i]);
    }
}

__global__ __launch_bounds__(256)
void attn_lepe_kernel(const u16* __restrict__ qkv,
                      const float* __restrict__ lepe_h,
                      const float* __restrict__ lepe_v,
                      u16* __restrict__ out) {
    const float rscale = 0.17677669529663687f; // 1/sqrt(32)

    int t = blockIdx.x * blockDim.x + threadIdx.x;   // < 2^19
    const int i     = t & 7;  t >>= 3;   // query row within stripe
    const int head  = t & 7;  t >>= 3;
    const int outer = t & 63; t >>= 6;   // w (axis0) or h (axis1)
    const int s     = t & 7;  t >>= 3;   // stripe index
    const int b     = t & 7;  t >>= 3;
    const int axis  = t;                 // 0 or 1

    int h, w, cb;
    const float* lw;
    if (axis == 0) { h = s * SS + i; w = outer; cb = head * HD;        lw = lepe_h; }
    else           { w = s * SS + i; h = outer; cb = HALF + head * HD; lw = lepe_v; }
    const int lch = head * HD;

    const size_t mrow = (size_t)b * LL + h * WW + w;

    float q[32];
    load_row32(qkv + mrow * NQKV + cb, q);

    size_t mj[SS];
    #pragma unroll
    for (int j = 0; j < SS; j++) {
        int hj = (axis == 0) ? (s * SS + j) : h;
        int wj = (axis == 0) ? w : (s * SS + j);
        mj[j] = (size_t)b * LL + hj * WW + wj;
    }

    float sc[SS];
    #pragma unroll
    for (int j = 0; j < SS; j++) {
        float kr[32];
        load_row32(qkv + mj[j] * NQKV + CC + cb, kr);
        float d = 0.0f;
        #pragma unroll
        for (int c = 0; c < 32; c++) d += q[c] * kr[c];
        sc[j] = d * rscale;
    }

    float mx = sc[0];
    #pragma unroll
    for (int j = 1; j < SS; j++) mx = fmaxf(mx, sc[j]);
    float sum = 0.0f;
    #pragma unroll
    for (int j = 0; j < SS; j++) { sc[j] = __expf(sc[j] - mx); sum += sc[j]; }
    const float inv = 1.0f / sum;

    float o[32];
    #pragma unroll
    for (int c = 0; c < 32; c++) o[c] = 0.0f;
    #pragma unroll
    for (int j = 0; j < SS; j++) {
        float vr[32];
        load_row32(qkv + mj[j] * NQKV + 2 * CC + cb, vr);
        const float p = sc[j] * inv;
        #pragma unroll
        for (int c = 0; c < 32; c++) o[c] += p * vr[c];
    }

    // LePE: depthwise 3x3 (SAME) on v at this pixel
    #pragma unroll
    for (int dy = 0; dy < 3; dy++) {
        const int hh = h + dy - 1;
        if (hh < 0 || hh >= HH) continue;
        #pragma unroll
        for (int dx = 0; dx < 3; dx++) {
            const int wp = w + dx - 1;
            if (wp < 0 || wp >= WW) continue;
            const size_t mp = (size_t)b * LL + hh * WW + wp;
            float vr[32];
            load_row32(qkv + mp * NQKV + 2 * CC + cb, vr);
            const float* wv = lw + (dy * 3 + dx) * HALF + lch;
            #pragma unroll
            for (int c = 0; c < 32; c++) o[c] += wv[c] * vr[c];
        }
    }

    ushort16 s0, s1;
    #pragma unroll
    for (int c = 0; c < 16; c++) { s0[c] = f2bf(o[c]); s1[c] = f2bf(o[16 + c]); }
    u16* op = out + mrow * CC + cb;
    *(ushort16*)(op)      = s0;
    *(ushort16*)(op + 16) = s1;
}

// ---------------- launch ----------------
extern "C" void kernel_launch(void* const* d_in, const int* in_sizes, int n_in,
                              void* d_out, int out_size, void* d_ws, size_t ws_size,
                              hipStream_t stream) {
    (void)in_sizes; (void)n_in; (void)out_size; (void)ws_size;

    const float* x      = (const float*)d_in[0];  // (8, 4096, 512)
    const float* w_qkv  = (const float*)d_in[1];  // (1536, 512)
    const float* w_proj = (const float*)d_in[2];  // (512, 512)
    const float* b_proj = (const float*)d_in[3];  // (512,)
    const float* lepe_h = (const float*)d_in[4];  // (3,3,1,256)
    const float* lepe_v = (const float*)d_in[5];  // (3,3,1,256)
    float* out = (float*)d_out;                   // (8, 4096, 512) fp32

    // workspace layout (bytes, 256-aligned)
    char* ws = (char*)d_ws;
    const size_t x_n     = (size_t)MTOT * CC;       // 16,777,216
    const size_t wqkv_n  = (size_t)NQKV * CC;       //    786,432
    const size_t wproj_n = (size_t)CC * CC;         //    262,144

    u16* x_bf = (u16*)ws;
    char* p1 = ws + ((x_n * 2 + 255) & ~255ull);
    u16* wqkv_bf = (u16*)p1;
    char* p2 = p1 + ((wqkv_n * 2 + 255) & ~255ull);
    u16* wproj_bf = (u16*)p2;
    char* p3 = p2 + ((wproj_n * 2 + 255) & ~255ull);
    u16* qkv_bf = (u16*)p3;                          // 32768*1536*2 = 100,663,296
    char* p4 = p3 + (((size_t)MTOT * NQKV * 2 + 255) & ~255ull);
    u16* attn_bf = (u16*)p4;                         // 32768*512*2  =  33,554,432

    // 1) x and weights -> bf16
    cvt_f32_bf16x8_kernel<<<(int)(x_n / 8 / 256), 256, 0, stream>>>(
        x, x_bf, (int)(x_n / 8));
    cvt_f32_bf16x8_kernel<<<(int)(wqkv_n / 8 / 256), 256, 0, stream>>>(
        w_qkv, wqkv_bf, (int)(wqkv_n / 8));
    cvt_f32_bf16x8_kernel<<<(int)(wproj_n / 8 / 256), 256, 0, stream>>>(
        w_proj, wproj_bf, (int)(wproj_n / 8));

    // 2) QKV GEMM: (32768x512 bf16) @ (512x1536) -> bf16
    {
        dim3 grid(MTOT / 64, NQKV / 128);
        wmma_gemm_async_kernel<true><<<grid, 256, 0, stream>>>(
            x_bf, wqkv_bf, nullptr, (void*)qkv_bf, MTOT, NQKV, CC);
    }

    // 3) stripe attention + LePE -> bf16 (32768x512)
    {
        const int total = 2 * BB * NS * 64 * NH * SS; // 524288
        attn_lepe_kernel<<<total / 256, 256, 0, stream>>>(
            qkv_bf, lepe_h, lepe_v, attn_bf);
    }

    // 4) proj GEMM + bias: (32768x512 bf16) @ (512x512) -> fp32 out
    {
        dim3 grid(MTOT / 64, CC / 128);
        wmma_gemm_async_kernel<false><<<grid, 256, 0, stream>>>(
            attn_bf, wproj_bf, b_proj, (void*)out, MTOT, CC, CC);
    }
}
